// DDRSA_ProbSparse_17600775979573
// MI455X (gfx1250) — matrix operand order repaired and checked by reading, not compiled, not run on hardware
//
#include <hip/hip_runtime.h>
#include <hip/hip_bf16.h>
#include <math.h>

// ---------------------------------------------------------------------------
// DDRSA ProbSparse Informer forward for MI455X (gfx1250, wave32, WMMA).
// All heavy GEMMs -> v_wmma_f32_16x16x32_f16 (f16 in, f32 accum).
// Weights are pre-packed TRANSPOSED (n-major, k-contiguous) so both A and B
// WMMA fragments are pure ds_load_b128 pairs from padded, bank-conflict-free
// LDS (row pitch 56 halfs = 112B). Workgroup tile 128x64, 4 WMMAs per A frag.
// LSTM decoder -> single persistent 32-wave workgroup, WMMA per step,
// LDS-resident state, ds_add_f32 for the output projection.
// Workspace requirement: ~272 MB.
// ---------------------------------------------------------------------------

typedef _Float16 v16h __attribute__((ext_vector_type(16)));
typedef _Float16 h8   __attribute__((ext_vector_type(8)));
typedef float    v8f  __attribute__((ext_vector_type(8)));

__device__ __forceinline__ float sigmoidf_(float x) { return 1.f / (1.f + expf(-x)); }

// ---- CDNA5 wave32 WMMA fragment gathers (contiguous k = 16*(lane>>4)+e, same in A and B) ----------
// A fragment 16x32 f16 from a row-major (k-contiguous) panel:
// lane r=lane&15 -> row; lanes 0-15 K=0..15, lanes 16-31 K=16..31.
// Elements 0..15 = K[16*(lane>>4) .. 16*(lane>>4)+15] (contiguous)
// -> exactly two 16-byte vector loads per lane.
__device__ __forceinline__ v16h load_A_frag(const _Float16* p0, int ldk) {
  int lane = threadIdx.x & 31;
  const _Float16* p = p0 + (size_t)(lane & 15) * ldk + ((lane >> 4) << 4);
  h8 lo = *(const h8*)p;
  h8 hi = *(const h8*)(p + 8);
  v16h a;
#pragma unroll
  for (int e = 0; e < 8; ++e) { a[e] = lo[e]; a[e + 8] = hi[e]; }
  return a;
}
// B fragment 32x16 f16 from a TRANSPOSED (n-major, k-contiguous) panel:
// lane c=lane&15 -> column; lanes 0-15 K=0..15, lanes 16-31 K=16..31,
// all 16 K-values contiguous -> two 16-byte vector loads per lane.
__device__ __forceinline__ v16h load_BT_frag(const _Float16* base, int colbase, int ldk) {
  int lane = threadIdx.x & 31;
  const _Float16* p = base + (size_t)(colbase + (lane & 15)) * ldk + ((lane >> 4) << 4);
  h8 lo = *(const h8*)p;
  h8 hi = *(const h8*)(p + 8);
  v16h b;
#pragma unroll
  for (int e = 0; e < 8; ++e) { b[e] = lo[e]; b[e + 8] = hi[e]; }
  return b;
}

// ---------------------------------------------------------------------------
// Generic tiled WMMA GEMM: C[M,N] = act(A[M,K] * BT[N,K]^T + bias (+ C if acc))
// B operand supplied TRANSPOSED (n-major). rowmap (optional) remaps tile rows
// to source rows of A (wrap-padded conv). ACT: 0 none, 1 exact GELU.
// K % 32 == 0, N % 64 == 0 required. Tile: 128x64, 8 waves, 4 WMMA / A-frag.
// ---------------------------------------------------------------------------
#define LDP 56  // LDS row pitch in halfs: 112B (16B aligned, conflict-free)

template <int ACT>
__global__ __launch_bounds__(256) void gemm_wmma(
    const _Float16* __restrict__ A, const _Float16* __restrict__ Bt,
    const float* __restrict__ bias, const int* __restrict__ rowmap,
    float* __restrict__ Cf, _Float16* __restrict__ Ch, int accumulate,
    int M, int N, int K) {
  __shared__ __align__(16) _Float16 As[128 * LDP];
  __shared__ __align__(16) _Float16 Bs[64 * LDP];
  const int tid = threadIdx.x;
  const int wave = tid >> 5, lane = tid & 31;
  const int wm = wave * 16;
  const int tileM = blockIdx.y * 128, tileN = blockIdx.x * 64;
  v8f acc[4] = {};

  for (int k0 = 0; k0 < K; k0 += 32) {
    __syncthreads();
    {  // stage A panel 128x32 (16 halfs / thread, k-contiguous)
      int r = tid >> 1, c = (tid & 1) * 16;
      int gr = tileM + r;
      h8 v0 = {}, v1 = {};
      if (gr < M) {
        int src = rowmap ? rowmap[gr] : gr;
        const _Float16* p = A + (size_t)src * K + k0 + c;
        v0 = *(const h8*)p;
        v1 = *(const h8*)(p + 8);
        if (k0 + 64 < K)  // prefetch next K panel -> global_prefetch_b8
          __builtin_prefetch(p + 32, 0, 1);
      }
      *(h8*)(&As[r * LDP + c]) = v0;
      *(h8*)(&As[r * LDP + c + 8]) = v1;
    }
    {  // stage BT panel 64x32 (8 halfs / thread, k-contiguous)
      int r = tid >> 2, c = (tid & 3) * 8;
      int gn = tileN + r;
      h8 v = {};
      if (gn < N) {
        const _Float16* p = Bt + (size_t)gn * K + k0 + c;
        v = *(const h8*)p;
        if (k0 + 64 < K)
          __builtin_prefetch(p + 32, 0, 1);
      }
      *(h8*)(&Bs[r * LDP + c]) = v;
    }
    __syncthreads();
    v16h a = load_A_frag(&As[wm * LDP], LDP);
#pragma unroll
    for (int s = 0; s < 4; ++s) {
      v16h b = load_BT_frag(&Bs[0], s * 16, LDP);
      acc[s] = __builtin_amdgcn_wmma_f32_16x16x32_f16(false, a, false, b, (short)0, acc[s], false, false);
    }
  }

  const int n = lane & 15, mo = (lane >> 4) * 8;
#pragma unroll
  for (int s = 0; s < 4; ++s) {
    int col = tileN + s * 16 + n;
    if (col >= N) continue;
    float bv = bias ? bias[col] : 0.f;
#pragma unroll
    for (int v = 0; v < 8; ++v) {
      int row = tileM + wm + mo + v;
      if (row >= M) continue;
      size_t o = (size_t)row * N + col;
      float val = acc[s][v] + bv;
      if (accumulate && Cf) val += Cf[o];
      if (ACT == 1) val = 0.5f * val * (1.f + erff(val * 0.7071067811865475f));
      if (Cf) Cf[o] = val;
      if (Ch) Ch[o] = (_Float16)val;
    }
  }
}

// ---------------------------------------------------------------------------
// Elementwise / reduction helpers
// ---------------------------------------------------------------------------
__global__ void cast_f16_kernel(const float* __restrict__ s, _Float16* __restrict__ d, int n) {
  int i = blockIdx.x * 256 + threadIdx.x;
  if (i < n) d[i] = (_Float16)s[i];
}

// batched transpose-cast: dst[lay][n][k] = src[lay][k][n]   (weights -> BT f16)
__global__ void bt_cast_kernel(const float* __restrict__ src, _Float16* __restrict__ dst,
                               int batch, int K, int N) {
  int i = blockIdx.x * 256 + threadIdx.x;
  int kn = K * N;
  if (i >= batch * kn) return;
  int lay = i / kn, r = i % kn, n = r / K, k = r % K;
  dst[i] = (_Float16)src[(size_t)lay * kn + (size_t)k * N + n];
}

__global__ void add_pe_kernel(float* __restrict__ h, _Float16* __restrict__ h16, int L, int total) {
  int i = blockIdx.x * 256 + threadIdx.x;
  if (i >= total) return;
  int j = i & 511, m = i >> 9, pos = m % L;
  int p = j >> 1;
  float freq = expf(-(float)(2 * p) * (9.210340371976184f / 512.f));
  float pe = (j & 1) ? cosf(pos * freq) : sinf(pos * freq);
  float v = h[i] + pe;
  h[i] = v;
  h16[i] = (_Float16)v;
}

__global__ void ln_residual_kernel(const float* __restrict__ x, const float* __restrict__ a,
                                   const float* __restrict__ g, const float* __restrict__ be,
                                   float* __restrict__ outf, _Float16* __restrict__ outh) {
  __shared__ float red[256];
  size_t base = (size_t)blockIdx.x * 512;
  int t = threadIdx.x;
  float v0 = x[base + t] + a[base + t];
  float v1 = x[base + 256 + t] + a[base + 256 + t];
  red[t] = v0 + v1;
  __syncthreads();
  for (int s = 128; s > 0; s >>= 1) { if (t < s) red[t] += red[t + s]; __syncthreads(); }
  float mean = red[0] * (1.f / 512.f);
  __syncthreads();
  float d0 = v0 - mean, d1 = v1 - mean;
  red[t] = d0 * d0 + d1 * d1;
  __syncthreads();
  for (int s = 128; s > 0; s >>= 1) { if (t < s) red[t] += red[t + s]; __syncthreads(); }
  float rstd = rsqrtf(red[0] * (1.f / 512.f) + 1e-5f);
  float o0 = d0 * rstd * g[t] + be[t];
  float o1 = d1 * rstd * g[t + 256] + be[t + 256];
  outf[base + t] = o0;         outh[base + t] = (_Float16)o0;
  outf[base + 256 + t] = o1;   outh[base + 256 + t] = (_Float16)o1;
}

// M = max(qk_sampled) - mean(qk_sampled over L)
__global__ void qk_m_kernel(const float* __restrict__ Q, const float* __restrict__ Kc,
                            const int* __restrict__ idx, float* __restrict__ Mout,
                            int Hn, int L, int u) {
  int g = blockIdx.x * 256 + threadIdx.x;
  if (g >= 8 * Hn * L) return;
  int l = g % L, bh = g / L, hh = bh % Hn, b = bh / Hn;
  const float* q = Q + ((size_t)(b * L + l) * 512 + hh * 64);
  float mx = -3.4e38f, sm = 0.f;
  for (int ui = 0; ui < u; ++ui) {
    int ki = idx[l * u + ui];
    const float* k = Kc + ((size_t)(b * L + ki) * 512 + hh * 64);
    float s = 0.f;
    for (int e = 0; e < 64; ++e) s += q[e] * k[e];
    mx = fmaxf(mx, s);
    sm += s;
  }
  Mout[(size_t)bh * L + l] = mx - sm / (float)L;
}

__global__ __launch_bounds__(256) void topk_kernel(const float* __restrict__ Mb,
                                                   int* __restrict__ mtop, int L, int u) {
  __shared__ float sval[256];
  __shared__ int sidx[256];
  __shared__ unsigned char mask[2048];
  int bh = blockIdx.x;
  const float* m = Mb + (size_t)bh * L;
  for (int i = threadIdx.x; i < L; i += 256) mask[i] = 0;
  __syncthreads();
  for (int it = 0; it < u; ++it) {
    float best = -3.4e38f;
    int bi = 0;
    for (int i = threadIdx.x; i < L; i += 256)
      if (!mask[i] && m[i] > best) { best = m[i]; bi = i; }
    sval[threadIdx.x] = best; sidx[threadIdx.x] = bi;
    __syncthreads();
    for (int s = 128; s > 0; s >>= 1) {
      if (threadIdx.x < s && sval[threadIdx.x + s] > sval[threadIdx.x]) {
        sval[threadIdx.x] = sval[threadIdx.x + s];
        sidx[threadIdx.x] = sidx[threadIdx.x + s];
      }
      __syncthreads();
    }
    if (threadIdx.x == 0) { mtop[bh * u + it] = sidx[0]; mask[sidx[0]] = 1; }
    __syncthreads();
  }
}

// scores -> softmax -> ctx for one (b, h, u_i)
__global__ __launch_bounds__(256) void attn_ctx_kernel(
    const float* __restrict__ Q, const float* __restrict__ Kc, const float* __restrict__ Vc,
    const int* __restrict__ mtop, float* __restrict__ ctx, int Hn, int L, int u) {
  __shared__ float sc[2048];
  __shared__ float red[256];
  __shared__ float qs[64];
  __shared__ float cpart[4][64];
  int blk = blockIdx.x;
  int ui = blk % u, bh = blk / u, hh = bh % Hn, b = bh / Hn;
  int mrow = mtop[bh * u + ui];
  const float* qp = Q + ((size_t)(b * L + mrow) * 512 + hh * 64);
  if (threadIdx.x < 64) qs[threadIdx.x] = qp[threadIdx.x];
  __syncthreads();
  float lmax = -3.4e38f;
  for (int l = threadIdx.x; l < L; l += 256) {
    const float* kp = Kc + ((size_t)(b * L + l) * 512 + hh * 64);
    float s = 0.f;
    for (int e = 0; e < 64; ++e) s += qs[e] * kp[e];
    s *= 0.125f;  // 1/sqrt(64)
    sc[l] = s;
    lmax = fmaxf(lmax, s);
  }
  red[threadIdx.x] = lmax;
  __syncthreads();
  for (int s = 128; s > 0; s >>= 1) { if (threadIdx.x < s) red[threadIdx.x] = fmaxf(red[threadIdx.x], red[threadIdx.x + s]); __syncthreads(); }
  float mx = red[0];
  __syncthreads();
  float lsum = 0.f;
  for (int l = threadIdx.x; l < L; l += 256) { float e = expf(sc[l] - mx); sc[l] = e; lsum += e; }
  red[threadIdx.x] = lsum;
  __syncthreads();
  for (int s = 128; s > 0; s >>= 1) { if (threadIdx.x < s) red[threadIdx.x] += red[threadIdx.x + s]; __syncthreads(); }
  float inv = 1.f / red[0];
  int dkk = threadIdx.x & 63, ch = threadIdx.x >> 6, seg = L >> 2;
  float a = 0.f;
  for (int l = ch * seg; l < (ch + 1) * seg; ++l)
    a += sc[l] * Vc[(size_t)(b * L + l) * 512 + hh * 64 + dkk];
  cpart[ch][dkk] = a;
  __syncthreads();
  if (threadIdx.x < 64)
    ctx[((size_t)bh * u + ui) * 64 + dkk] =
        (cpart[0][dkk] + cpart[1][dkk] + cpart[2][dkk] + cpart[3][dkk]) * inv;
}

__global__ __launch_bounds__(256) void vmean_kernel(const float* __restrict__ V,
                                                    float* __restrict__ vm, int Hn, int L) {
  __shared__ float part[4][64];
  int bh = blockIdx.x, hh = bh % Hn, b = bh / Hn;
  int dkk = threadIdx.x & 63, ch = threadIdx.x >> 6, seg = L >> 2;
  float s = 0.f;
  for (int l = ch * seg; l < (ch + 1) * seg; ++l)
    s += V[(size_t)(b * L + l) * 512 + hh * 64 + dkk];
  part[ch][dkk] = s;
  __syncthreads();
  if (threadIdx.x < 64)
    vm[bh * 64 + dkk] = (part[0][dkk] + part[1][dkk] + part[2][dkk] + part[3][dkk]) / (float)L;
}

__global__ void fill_ctx_kernel(const float* __restrict__ vm, float* __restrict__ ctxo, int L, int total) {
  int i = blockIdx.x * 256 + threadIdx.x;
  if (i >= total) return;
  int dd = i & 511;
  int b = i / (512 * L);
  ctxo[i] = vm[b * 512 + dd];
}

__global__ void scatter_ctx_kernel(const float* __restrict__ ctx, const int* __restrict__ mtop,
                                   float* __restrict__ ctxo, int Hn, int L, int u) {
  int i = blockIdx.x * 256 + threadIdx.x;
  if (i >= 64 * u * 64) return;
  int dkk = i & 63, r = i >> 6;
  int ui = r % u, bh = r / u, hh = bh % Hn, b = bh / Hn;
  int l = mtop[bh * u + ui];
  ctxo[(size_t)(b * L + l) * 512 + hh * 64 + dkk] = ctx[i];
}

__global__ void bn_elu_kernel(const float* __restrict__ y, float* __restrict__ o,
                              const float* __restrict__ g, const float* __restrict__ b,
                              const float* __restrict__ m, const float* __restrict__ v, int total) {
  int i = blockIdx.x * 256 + threadIdx.x;
  if (i >= total) return;
  int c = i & 511;
  float t = (y[i] - m[c]) * rsqrtf(v[c] + 1e-5f) * g[c] + b[c];
  o[i] = t > 0.f ? t : expf(t) - 1.f;
}

__global__ void pool_kernel(const float* __restrict__ y, float* __restrict__ outf,
                            _Float16* __restrict__ outh, int Lin, int Lout, int total) {
  int i = blockIdx.x * 256 + threadIdx.x;
  if (i >= total) return;
  int c = i & 511, r = i >> 9, to = r % Lout, b = r / Lout;
  float m = -3.4e38f;
  for (int k = -1; k <= 1; ++k) {
    int ti = 2 * to + k;
    if (ti >= 0 && ti < Lin) m = fmaxf(m, y[(size_t)(b * Lin + ti) * 512 + c]);
  }
  outf[i] = m;
  outh[i] = (_Float16)m;
}

__global__ void mean_kernel(const float* __restrict__ x, float* __restrict__ enc,
                            _Float16* __restrict__ enc16, int L) {
  int i = blockIdx.x * 256 + threadIdx.x;
  if (i >= 8 * 512) return;
  int b = i >> 9, dd = i & 511;
  float s = 0.f;
  for (int l = 0; l < L; ++l) s += x[(size_t)(b * L + l) * 512 + dd];
  s /= (float)L;
  enc[i] = s;
  enc16[i] = (_Float16)s;
}

__global__ void rowmap_kernel(int* __restrict__ rm, int L) {
  int i = blockIdx.x * 256 + threadIdx.x;
  if (i >= 3 * 8 * 2048) return;
  int k = i / (8 * 2048), m = i % (8 * 2048), b = m / L, t = m % L;
  rm[i] = b * L + ((t + k - 1 + L) % L);
}

// conv weights, transposed layout: pw[k][o][ic] = conv_w[o][ic][k]  (BT_k, ldb=512)
__global__ void pack_convw_kernel(const float* __restrict__ cw, _Float16* __restrict__ pw) {
  int i = blockIdx.x * 256 + threadIdx.x;
  if (i >= 3 * 512 * 512) return;
  int k = i / (512 * 512), r = i % (512 * 512), o = r / 512, ic = r % 512;
  pw[i] = (_Float16)cw[(o * 512 + ic) * 3 + k];
}

__global__ void biascomb_kernel(const float* __restrict__ a, const float* __restrict__ b,
                                float* __restrict__ o, int n) {
  int i = blockIdx.x * 256 + threadIdx.x;
  if (i < n) o[i] = a[i] + b[i];
}

// ---------------------------------------------------------------------------
// Persistent LSTM decoder: one 32-wave workgroup, 100 serial steps.
// Each step: gates(8x2048) = gx + h(8x512) @ whh^T via 64 WMMAs,
// cell state in registers, h in LDS, output projection via ds_add_f32.
// whh f16 is already n-major (2048x512) = BT layout, streamed from L2.
// ---------------------------------------------------------------------------
__global__ __launch_bounds__(1024) void lstm_wmma_kernel(
    const float* __restrict__ gx, const _Float16* __restrict__ whh16,
    const float* __restrict__ out_w, const float* __restrict__ out_b,
    float* __restrict__ out, int T) {
  __shared__ __align__(16) _Float16 hS[16 * 512];   // h (rows 8..15 stay zero)
  __shared__ _Float16 gates[8 * 2048];
  __shared__ float outacc[8];
  const int tid = threadIdx.x, wave = tid >> 5, lane = tid & 31;
  for (int i = tid; i < 16 * 512; i += 1024) hS[i] = (_Float16)0;
  if (tid < 8) outacc[tid] = 0.f;
  float creg[4] = {0.f, 0.f, 0.f, 0.f};
  __syncthreads();
  const int nbase = wave * 64;
  const int n = lane & 15, mo = (lane >> 4) * 8;
  const float ob = out_b[0];

  for (int t = 0; t < T; ++t) {
    v8f acc[4] = {};
    for (int k0 = 0; k0 < 512; k0 += 32) {
      v16h a = load_A_frag(hS + k0, 512);
#pragma unroll
      for (int s = 0; s < 4; ++s) {
        v16h b = load_BT_frag(whh16 + k0, nbase + s * 16, 512);
        acc[s] = __builtin_amdgcn_wmma_f32_16x16x32_f16(false, a, false, b, (short)0, acc[s], false, false);
      }
    }
#pragma unroll
    for (int s = 0; s < 4; ++s) {
      int col = nbase + s * 16 + n;
#pragma unroll
      for (int v = 0; v < 8; ++v) {
        int row = mo + v;
        if (row < 8) gates[row * 2048 + col] = (_Float16)(acc[s][v] + gx[row * 2048 + col]);
      }
    }
    __syncthreads();
#pragma unroll
    for (int p = 0; p < 4; ++p) {
      int idx = tid + p * 1024;
      int b = idx >> 9, j = idx & 511;
      float ig = sigmoidf_((float)gates[b * 2048 + j]);
      float fg = sigmoidf_((float)gates[b * 2048 + 512 + j]);
      float gg = tanhf((float)gates[b * 2048 + 1024 + j]);
      float og = sigmoidf_((float)gates[b * 2048 + 1536 + j]);
      creg[p] = fg * creg[p] + ig * gg;
      float hn = og * tanhf(creg[p]);
      hS[b * 512 + j] = (_Float16)hn;
      atomicAdd(&outacc[b], hn * out_w[j]);
    }
    __syncthreads();
    if (tid < 8) {
      out[tid * T + t] = outacc[tid] + ob;
      outacc[tid] = 0.f;
    }
  }
}

// ---------------------------------------------------------------------------
// Host side
// ---------------------------------------------------------------------------
static void gemm(hipStream_t st, const _Float16* A, const _Float16* Bt, const float* bias,
                 const int* rowmap, float* Cf, _Float16* Ch, int acc,
                 int M, int N, int K, int act) {
  dim3 grid(N / 64, (M + 127) / 128);
  if (act) gemm_wmma<1><<<grid, 256, 0, st>>>(A, Bt, bias, rowmap, Cf, Ch, acc, M, N, K);
  else     gemm_wmma<0><<<grid, 256, 0, st>>>(A, Bt, bias, rowmap, Cf, Ch, acc, M, N, K);
}

static inline dim3 eg(int n) { return dim3((n + 255) / 256); }

static void encoder_layer(hipStream_t st, const float* xf, const _Float16* xh, int L, int u,
                          const int* idx,
                          const _Float16* wq16, const float* bq, const _Float16* wk16, const float* bk,
                          const _Float16* wv16, const float* bv, const _Float16* wo16, const float* bo,
                          const _Float16* w116, const float* b1, const _Float16* w216, const float* b2,
                          const float* g1, const float* be1, const float* g2, const float* be2,
                          float* Qb, float* Kb, float* Vb, _Float16* ctx16, _Float16* ln1h,
                          _Float16* ffn16, float* outf, _Float16* outh,
                          float* Mbuf, int* mtop, float* ctxin, float* vmean) {
  const int M = 8 * L;
  gemm(st, xh, wq16, bq, nullptr, Qb, nullptr, 0, M, 512, 512, 0);
  gemm(st, xh, wk16, bk, nullptr, Kb, nullptr, 0, M, 512, 512, 0);
  gemm(st, xh, wv16, bv, nullptr, Vb, nullptr, 0, M, 512, 512, 0);
  qk_m_kernel<<<eg(8 * 8 * L), 256, 0, st>>>(Qb, Kb, idx, Mbuf, 8, L, u);
  topk_kernel<<<64, 256, 0, st>>>(Mbuf, mtop, L, u);
  attn_ctx_kernel<<<64 * u, 256, 0, st>>>(Qb, Kb, Vb, mtop, ctxin, 8, L, u);
  vmean_kernel<<<64, 256, 0, st>>>(Vb, vmean, 8, L);
  fill_ctx_kernel<<<eg(M * 512), 256, 0, st>>>(vmean, Qb, L, M * 512);
  scatter_ctx_kernel<<<eg(64 * u * 64), 256, 0, st>>>(ctxin, mtop, Qb, 8, L, u);
  cast_f16_kernel<<<eg(M * 512), 256, 0, st>>>(Qb, ctx16, M * 512);
  gemm(st, ctx16, wo16, bo, nullptr, Kb, nullptr, 0, M, 512, 512, 0);
  ln_residual_kernel<<<M, 256, 0, st>>>(xf, Kb, g1, be1, Vb, ln1h);
  gemm(st, ln1h, w116, b1, nullptr, nullptr, ffn16, 0, M, 2048, 512, 1);   // +GELU, f16-only out
  gemm(st, ffn16, w216, b2, nullptr, Qb, nullptr, 0, M, 512, 2048, 0);
  ln_residual_kernel<<<M, 256, 0, st>>>(Vb, Qb, g2, be2, outf, outh);
}

#define MB_ (1024ull * 1024ull)

extern "C" void kernel_launch(void* const* d_in, const int* in_sizes, int n_in,
                              void* d_out, int out_size, void* d_ws, size_t ws_size,
                              hipStream_t stream) {
  (void)in_sizes; (void)n_in; (void)out_size; (void)ws_size;
  const float* x      = (const float*)d_in[0];
  const int*   idx1   = (const int*)d_in[1];
  const int*   idx2   = (const int*)d_in[2];
  const float* emb_w  = (const float*)d_in[3];
  const float* emb_b  = (const float*)d_in[4];
  const float* wq     = (const float*)d_in[5];
  const float* bq     = (const float*)d_in[6];
  const float* wk     = (const float*)d_in[7];
  const float* bk     = (const float*)d_in[8];
  const float* wv     = (const float*)d_in[9];
  const float* bv     = (const float*)d_in[10];
  const float* wo     = (const float*)d_in[11];
  const float* bo     = (const float*)d_in[12];
  const float* w1     = (const float*)d_in[13];
  const float* b1     = (const float*)d_in[14];
  const float* w2     = (const float*)d_in[15];
  const float* b2     = (const float*)d_in[16];
  const float* ln1g   = (const float*)d_in[17];
  const float* ln1b   = (const float*)d_in[18];
  const float* ln2g   = (const float*)d_in[19];
  const float* ln2b   = (const float*)d_in[20];
  const float* conv_w = (const float*)d_in[21];
  const float* conv_b = (const float*)d_in[22];
  const float* bn_g   = (const float*)d_in[23];
  const float* bn_b   = (const float*)d_in[24];
  const float* bn_m   = (const float*)d_in[25];
  const float* bn_v   = (const float*)d_in[26];
  const float* wih    = (const float*)d_in[27];
  const float* whh    = (const float*)d_in[28];
  const float* bih    = (const float*)d_in[29];
  const float* bhh    = (const float*)d_in[30];
  const float* outw   = (const float*)d_in[31];
  const float* outb   = (const float*)d_in[32];
  float* out = (float*)d_out;

  // ---- workspace layout (~272 MB) ----
  char* W = (char*)d_ws;
  _Float16* X16  = (_Float16*)(W + 0);
  float* hA      = (float*)(W + 4 * MB_);
  float* hB      = (float*)(W + 36 * MB_);
  float* hC      = (float*)(W + 68 * MB_);
  float* hD      = (float*)(W + 100 * MB_);
  _Float16* H16A = (_Float16*)(W + 132 * MB_);
  _Float16* H16B = (_Float16*)(W + 148 * MB_);
  _Float16* H16C = (_Float16*)(W + 164 * MB_);  // reserved
  (void)H16C;
  _Float16* FFN16 = (_Float16*)(W + 180 * MB_);
  _Float16* W16  = (_Float16*)(W + 244 * MB_);
  _Float16* wq16 = W16;                         // all weight blocks are BT (n-major)
  _Float16* wk16 = wq16 + 524288;
  _Float16* wv16 = wk16 + 524288;
  _Float16* wo16 = wv16 + 524288;
  _Float16* w116 = wo16 + 524288;
  _Float16* w216 = w116 + 2097152;
  _Float16* emb16 = w216 + 2097152;
  _Float16* packw = emb16 + 32768;
  _Float16* wiht  = packw + 786432;
  _Float16* whh16 = wiht + 1048576;
  char* S = W + 264 * MB_;
  float* Mbuf     = (float*)(S);
  int*   mtop     = (int*)(S + 1 * MB_);
  float* ctxin    = (float*)(S + 2 * MB_);
  float* vmean    = (float*)(S + 3 * MB_);
  int*   rowmap   = (int*)(S + 3 * MB_ + 64 * 1024);
  float* enc      = (float*)(S + 4 * MB_);
  _Float16* enc16 = (_Float16*)(S + 4 * MB_ + 64 * 1024);
  float* gx       = (float*)(S + 5 * MB_);
  float* biascomb = (float*)(S + 5 * MB_ + 256 * 1024);

  // ---- weight prep: transposed (BT) f16 packs, recomputed per call ----
  bt_cast_kernel<<<eg(524288), 256, 0, stream>>>(wq, wq16, 2, 512, 512);
  bt_cast_kernel<<<eg(524288), 256, 0, stream>>>(wk, wk16, 2, 512, 512);
  bt_cast_kernel<<<eg(524288), 256, 0, stream>>>(wv, wv16, 2, 512, 512);
  bt_cast_kernel<<<eg(524288), 256, 0, stream>>>(wo, wo16, 2, 512, 512);
  bt_cast_kernel<<<eg(2097152), 256, 0, stream>>>(w1, w116, 2, 512, 2048);
  bt_cast_kernel<<<eg(2097152), 256, 0, stream>>>(w2, w216, 2, 2048, 512);
  bt_cast_kernel<<<eg(32768), 256, 0, stream>>>(emb_w, emb16, 1, 64, 512);
  cast_f16_kernel<<<eg(1048576), 256, 0, stream>>>(whh, whh16, 1048576);  // (2048,512) = BT already
  cast_f16_kernel<<<eg(1048576), 256, 0, stream>>>(wih, wiht, 1048576);   // BT of wih^T = wih itself
  pack_convw_kernel<<<eg(786432), 256, 0, stream>>>(conv_w, packw);
  biascomb_kernel<<<eg(2048), 256, 0, stream>>>(bih, bhh, biascomb, 2048);
  rowmap_kernel<<<eg(3 * 16384), 256, 0, stream>>>(rowmap, 2048);

  // ---- embedding + positional encoding ----
  cast_f16_kernel<<<eg(1048576), 256, 0, stream>>>(x, X16, 1048576);     // (8,2048,64)
  gemm(stream, X16, emb16, emb_b, nullptr, hA, nullptr, 0, 16384, 512, 64, 0);
  add_pe_kernel<<<eg(16384 * 512), 256, 0, stream>>>(hA, H16A, 2048, 16384 * 512);

  // ---- encoder layer 1 (L=2048, u=40); out: f32 hC, f16 H16A ----
  encoder_layer(stream, hA, H16A, 2048, 40, idx1,
                wq16, bq, wk16, bk, wv16, bv, wo16, bo,
                w116, b1, w216, b2, ln1g, ln1b, ln2g, ln2b,
                hB, hC, hD, H16B, H16B, FFN16, hC, H16A,
                Mbuf, mtop, ctxin, vmean);

  // ---- conv distill = 3 accumulating row-remapped GEMMs + BN/ELU + maxpool ----
  gemm(stream, H16A, packw,          conv_b,  rowmap,          hA, nullptr, 0, 16384, 512, 512, 0);
  gemm(stream, H16A, packw + 262144, nullptr, rowmap + 16384,  hA, nullptr, 1, 16384, 512, 512, 0);
  gemm(stream, H16A, packw + 524288, nullptr, rowmap + 32768,  hA, nullptr, 1, 16384, 512, 512, 0);
  bn_elu_kernel<<<eg(16384 * 512), 256, 0, stream>>>(hA, hB, bn_g, bn_b, bn_m, bn_v, 16384 * 512);
  pool_kernel<<<eg(8 * 1024 * 512), 256, 0, stream>>>(hB, hD, H16B, 2048, 1024, 8 * 1024 * 512);

  // ---- encoder layer 2 (L=1024, u=35); out: f32 hB, f16 H16A ----
  encoder_layer(stream, hD, H16B, 1024, 35, idx2,
                wq16 + 262144, bq + 512, wk16 + 262144, bk + 512,
                wv16 + 262144, bv + 512, wo16 + 262144, bo + 512,
                w116 + 1048576, b1 + 2048, w216 + 1048576, b2 + 512,
                ln1g + 512, ln1b + 512, ln2g + 512, ln2b + 512,
                hA, hB, hC, H16A, H16A, FFN16, hB, H16A,
                Mbuf, mtop, ctxin, vmean);

  // ---- encoder mean -> LSTM decoder ----
  mean_kernel<<<eg(8 * 512), 256, 0, stream>>>(hB, enc, enc16, 1024);
  gemm(stream, enc16, wiht, biascomb, nullptr, gx, nullptr, 0, 8, 2048, 512, 0);  // gx = enc@wih^T + bih + bhh
  lstm_wmma_kernel<<<1, 1024, 0, stream>>>(gx, whh16, outw, outb, out, 100);
}
